// HIHI2_27393301414301
// MI455X (gfx1250) — matrix operand order repaired and checked by reading, not compiled, not run on hardware
//
#include <hip/hip_runtime.h>

// ---------------------------------------------------------------------------
// Residual VQ (3 stages) for MI455X / gfx1250, bf16 WMMA core.
// Shapes fixed by the reference: B=8, C=384, H=W=64, K=2048, N=B*H*W=32768.
// Async double-buffered global->LDS codebook staging in k_argmin.
// ---------------------------------------------------------------------------

typedef __attribute__((ext_vector_type(16))) __bf16 v16bf;
typedef __attribute__((ext_vector_type(8)))  float  v8f;
typedef __attribute__((__vector_size__(16))) int    v4i;

#define NPIX 32768      // B*H*W
#define CCH  384        // channels
#define KCB  2048       // codebook size
#define HWSZ 4096       // H*W

#if __has_builtin(__builtin_amdgcn_global_load_async_to_lds_b128)
#define HAVE_ASYNC_LDS 1
typedef __attribute__((address_space(1))) v4i* gv4i_p;   // global v4i*
typedef __attribute__((address_space(3))) v4i* lv4i_p;   // LDS v4i*
#else
#define HAVE_ASYNC_LDS 0
#endif

// round-to-nearest-even f32 -> bf16 (bit manipulation; no __bf16 cast reliance)
__device__ __forceinline__ __bf16 f2bf(float x) {
    union { float f; unsigned u; } v; v.f = x;
    unsigned r = v.u + 0x7FFFu + ((v.u >> 16) & 1u);
    union { unsigned short s; __bf16 b; } o; o.s = (unsigned short)(r >> 16);
    return o.b;
}

// ------------------------- prep kernels ------------------------------------

__global__ void k_zero_losses(float* loss) {
    if (threadIdx.x == 0) { loss[0] = 0.f; loss[1] = 0.f; loss[2] = 0.f; }
}

__global__ __launch_bounds__(256) void k_conv_w(const float* __restrict__ w,
                                                __bf16* __restrict__ wbf) {
    int i = blockIdx.x * 256 + threadIdx.x;      // 384*384 = 147456 elems
    wbf[i] = f2bf(w[i]);
}

// one block per codebook row: convert to bf16 and compute ||c||^2 in f32
__global__ __launch_bounds__(128) void k_prep_code(const float* __restrict__ v1,
                                                   const float* __restrict__ v2,
                                                   const float* __restrict__ v3,
                                                   __bf16* __restrict__ codebf,
                                                   float*  __restrict__ cn) {
    __shared__ float red[4];
    int bid = blockIdx.x;                 // 0 .. 3*2048-1
    int s = bid >> 11;                    // stage
    int k = bid & (KCB - 1);
    const float* src = (s == 0) ? v1 : (s == 1) ? v2 : v3;
    float sum = 0.f;
    for (int j = threadIdx.x; j < CCH; j += 128) {
        float x = src[(size_t)k * CCH + j];
        codebf[((size_t)s * KCB + k) * CCH + j] = f2bf(x);
        sum += x * x;
    }
    // wave reduce (wave32)
    #pragma unroll
    for (int off = 16; off >= 1; off >>= 1) sum += __shfl_xor(sum, off, 32);
    int lane = threadIdx.x & 31, wave = threadIdx.x >> 5;
    if (lane == 0) red[wave] = sum;
    __syncthreads();
    if (threadIdx.x == 0)
        cn[(size_t)s * KCB + k] = red[0] + red[1] + red[2] + red[3];
}

// ------------------------- projection GEMM ---------------------------------
// head(N x C) = feat_flat(N x C) @ w_proj^T + bias ; write f32 + bf16 mirrors.
__global__ __launch_bounds__(256) void k_proj(const float* __restrict__ feat,
                                              const float* __restrict__ bias,
                                              const __bf16* __restrict__ wbf,
                                              float* __restrict__ fbuf,
                                              __bf16* __restrict__ fbf) {
    int wave = threadIdx.x >> 5;
    int lane = threadIdx.x & 31;
    int rowBase = (blockIdx.x * 8 + wave) * 16;
    int col16 = lane & 15;
    int khalf = lane >> 4;
    int row = rowBase + col16;            // A-matrix row held by this lane
    int bimg = row >> 12;                 // image batch index
    int hw = row & (HWSZ - 1);

    // A fragments: lane holds row, K = kb + khalf*8 + i (elems 0..7)
    //                           K = kb + 16 + khalf*8 + i (elems 8..15)
    v16bf a[12];
    #pragma unroll
    for (int ks = 0; ks < 12; ++ks) {
        int kb = ks * 32;
        #pragma unroll
        for (int i = 0; i < 8; ++i) {
            int c0 = kb + khalf * 8 + i;
            a[ks][i]     = f2bf(feat[((size_t)bimg * CCH + c0)      * HWSZ + hw]);
            a[ks][8 + i] = f2bf(feat[((size_t)bimg * CCH + c0 + 16) * HWSZ + hw]);
        }
    }

    for (int ct = 0; ct < 24; ++ct) {
        int col = ct * 16 + col16;
        v8f acc = {};
        #pragma unroll
        for (int ks = 0; ks < 12; ++ks) {
            // B frag: elems 0..7 = w_proj[col, ks*32 + khalf*8 + i], elems 8..15 = +16
            const __bf16* bp = wbf + (size_t)col * CCH + ks * 32 + khalf * 8;
            union { uint4 u[2]; v16bf v; } t;
            t.u[0] = *(const uint4*)(bp);
            t.u[1] = *(const uint4*)(bp + 16);
            acc = __builtin_amdgcn_wmma_f32_16x16x32_bf16(
                false, a[ks], false, t.v, (short)0, acc, false, false);
        }
        float bv = bias[col];
        #pragma unroll
        for (int r = 0; r < 8; ++r) {
            int rr = r + khalf * 8;
            float v = acc[r] + bv;
            size_t o = (size_t)(rowBase + rr) * CCH + col;
            fbuf[o] = v;
            fbf[o]  = f2bf(v);
        }
    }
}

// ------------------------- distance argmin ---------------------------------
// For 16 rows per wave: argmin_k ( ||c_k||^2 - 2 f.c_k ) over K=2048 codes.
// Code tiles (16 codes x 384 ch bf16) double-buffered in LDS via the CDNA5
// async global->LDS path when available; shared by 8 waves.
__global__ __launch_bounds__(256) void k_argmin(const __bf16* __restrict__ fbf,
                                                const __bf16* __restrict__ codebf,
                                                const float*  __restrict__ cn,
                                                int* __restrict__ idxout) {
    __shared__ __align__(16) unsigned char lds[2][16 * 784];  // 784B padded rows
    int wave = threadIdx.x >> 5;
    int lane = threadIdx.x & 31;
    int rowBase = (blockIdx.x * 8 + wave) * 16;
    int col16 = lane & 15;
    int khalf = lane >> 4;
    int row = rowBase + col16;

    // this thread's 3 staging chunks (16B each) of a 16x768B code tile
    int j0 = threadIdx.x;             // chunk ids j0, j0+256, j0+512 of 768
    // preload A fragments for the whole K dimension (12 x v16bf)
    v16bf a[12];
    #pragma unroll
    for (int ks = 0; ks < 12; ++ks) {
        const __bf16* ap = fbf + (size_t)row * CCH + ks * 32 + khalf * 16;
        union { uint4 u[2]; v16bf v; } t;
        t.u[0] = *(const uint4*)(ap);
        t.u[1] = *(const uint4*)(ap + 8);    // +8 channels
        a[ks] = t.v;
    }

    float minv[8]; int mini[8];
    #pragma unroll
    for (int r = 0; r < 8; ++r) { minv[r] = 3.4e38f; mini[r] = 0; }

#if HAVE_ASYNC_LDS
    // async double-buffered staging: LDS fill of tile ct+1 overlaps WMMAs of ct
    #define ISSUE_TILE(CT, BUF)                                                 \
        {                                                                       \
            _Pragma("unroll")                                                   \
            for (int rep = 0; rep < 3; ++rep) {                                 \
                int j = j0 + rep * 256;                                         \
                int crow = j / 48, cchunk = j - crow * 48;                      \
                const uint4* gsrc =                                             \
                    (const uint4*)(codebf + ((size_t)((CT) * 16 + crow)) * CCH) \
                    + cchunk;                                                   \
                __builtin_amdgcn_global_load_async_to_lds_b128(                 \
                    (gv4i_p)gsrc,                                               \
                    (lv4i_p)(lds[BUF] + crow * 784 + cchunk * 16),              \
                    0, 0);                                                      \
            }                                                                   \
        }
    ISSUE_TILE(0, 0)
    for (int ct = 0; ct < KCB / 16; ++ct) {
        int cur = ct & 1;
        if (ct + 1 < KCB / 16) {
            ISSUE_TILE(ct + 1, cur ^ 1)
#if __has_builtin(__builtin_amdgcn_s_wait_asynccnt)
            __builtin_amdgcn_s_wait_asynccnt(3);   // tile ct done, ct+1 in flight
#else
            asm volatile("s_wait_asynccnt 0x3" ::: "memory");
#endif
        } else {
#if __has_builtin(__builtin_amdgcn_s_wait_asynccnt)
            __builtin_amdgcn_s_wait_asynccnt(0);
#else
            asm volatile("s_wait_asynccnt 0x0" ::: "memory");
#endif
        }
        __syncthreads();                           // all waves' fills visible

        v8f acc = {};
        #pragma unroll
        for (int ks = 0; ks < 12; ++ks) {
            const unsigned char* bp =
                lds[cur] + col16 * 784 + (ks * 32 + khalf * 16) * 2;
            union { uint4 u[2]; v16bf v; } t;
            t.u[0] = *(const uint4*)(bp);
            t.u[1] = *(const uint4*)(bp + 16);
            acc = __builtin_amdgcn_wmma_f32_16x16x32_bf16(
                false, a[ks], false, t.v, (short)0, acc, false, false);
        }
        int code = ct * 16 + col16;
        float c2 = cn[code];
        #pragma unroll
        for (int r = 0; r < 8; ++r) {
            float s = c2 - 2.0f * acc[r];
            if (s < minv[r]) { minv[r] = s; mini[r] = code; }
        }
        __syncthreads();     // all reads of lds[cur] done before it is refilled
    }
    #undef ISSUE_TILE
#else
    // synchronous fallback staging
    for (int ct = 0; ct < KCB / 16; ++ct) {
        __syncthreads();
        #pragma unroll
        for (int rep = 0; rep < 3; ++rep) {
            int j = j0 + rep * 256;
            int crow = j / 48, cchunk = j - crow * 48;
            const uint4* src =
                (const uint4*)(codebf + ((size_t)(ct * 16 + crow)) * CCH) + cchunk;
            *(uint4*)(lds[0] + crow * 784 + cchunk * 16) = *src;
        }
        __syncthreads();

        v8f acc = {};
        #pragma unroll
        for (int ks = 0; ks < 12; ++ks) {
            const unsigned char* bp =
                lds[0] + col16 * 784 + (ks * 32 + khalf * 16) * 2;
            union { uint4 u[2]; v16bf v; } t;
            t.u[0] = *(const uint4*)(bp);
            t.u[1] = *(const uint4*)(bp + 16);
            acc = __builtin_amdgcn_wmma_f32_16x16x32_bf16(
                false, a[ks], false, t.v, (short)0, acc, false, false);
        }
        int code = ct * 16 + col16;
        float c2 = cn[code];
        #pragma unroll
        for (int r = 0; r < 8; ++r) {
            float s = c2 - 2.0f * acc[r];
            if (s < minv[r]) { minv[r] = s; mini[r] = code; }
        }
    }
#endif

    // reduce across the 16 columns (stay within each 16-lane half)
    #pragma unroll
    for (int off = 8; off >= 1; off >>= 1) {
        #pragma unroll
        for (int r = 0; r < 8; ++r) {
            float ov = __shfl_xor(minv[r], off, 32);
            int   oi = __shfl_xor(mini[r], off, 32);
            if (ov < minv[r] || (ov == minv[r] && oi < mini[r])) {
                minv[r] = ov; mini[r] = oi;
            }
        }
    }
    if (col16 == 0) {
        #pragma unroll
        for (int r = 0; r < 8; ++r)
            idxout[rowBase + r + khalf * 8] = mini[r];
    }
}

// ------------------------- residual / gather / loss ------------------------
// q = code_f32[idx[n]]; r = f - q; f <- r (f32 + bf16); out (+)= q; loss += r^2
__global__ __launch_bounds__(256) void k_residual(float* __restrict__ fbuf,
                                                  __bf16* __restrict__ fbf,
                                                  const float* __restrict__ codef,
                                                  const int* __restrict__ idx,
                                                  float* __restrict__ out,
                                                  float* __restrict__ loss,
                                                  int firstStage) {
    __shared__ float red[8];
    int tid = blockIdx.x * 256 + threadIdx.x;    // total N * (C/4) = N*96
    int n  = tid / 96;
    int c4 = (tid - n * 96) * 4;
    int k = idx[n];
    const float4 q = *(const float4*)(codef + (size_t)k * CCH + c4);
    float4 f = *(const float4*)(fbuf + (size_t)n * CCH + c4);
    float4 r; r.x = f.x - q.x; r.y = f.y - q.y; r.z = f.z - q.z; r.w = f.w - q.w;
    *(float4*)(fbuf + (size_t)n * CCH + c4) = r;
    __bf16* fb = fbf + (size_t)n * CCH + c4;
    fb[0] = f2bf(r.x); fb[1] = f2bf(r.y); fb[2] = f2bf(r.z); fb[3] = f2bf(r.w);

    // output is (B, C, H, W)
    int bimg = n >> 12, hw = n & (HWSZ - 1);
    size_t ob = ((size_t)bimg * CCH + c4) * HWSZ + hw;
    if (firstStage) {
        out[ob]            = q.x;
        out[ob + HWSZ]     = q.y;
        out[ob + 2 * HWSZ] = q.z;
        out[ob + 3 * HWSZ] = q.w;
    } else {
        out[ob]            += q.x;
        out[ob + HWSZ]     += q.y;
        out[ob + 2 * HWSZ] += q.z;
        out[ob + 3 * HWSZ] += q.w;
    }

    float ls = r.x * r.x + r.y * r.y + r.z * r.z + r.w * r.w;
    #pragma unroll
    for (int off = 16; off >= 1; off >>= 1) ls += __shfl_xor(ls, off, 32);
    int lane = threadIdx.x & 31, wave = threadIdx.x >> 5;
    if (lane == 0) red[wave] = ls;
    __syncthreads();
    if (threadIdx.x == 0) {
        float s = 0.f;
        #pragma unroll
        for (int w = 0; w < 8; ++w) s += red[w];
        atomicAdd(loss, s);
    }
}

__global__ void k_final(const float* __restrict__ loss, float* __restrict__ out_loss) {
    if (threadIdx.x == 0)
        out_loss[0] = 1.25f * (loss[0] + loss[1] + loss[2]) /
                      (float)((size_t)NPIX * CCH);
}

// ------------------------- launcher ----------------------------------------

extern "C" void kernel_launch(void* const* d_in, const int* in_sizes, int n_in,
                              void* d_out, int out_size, void* d_ws, size_t ws_size,
                              hipStream_t stream) {
    const float* feat  = (const float*)d_in[0];   // (8,384,64,64)
    const float* wproj = (const float*)d_in[1];   // (384,384)
    const float* bproj = (const float*)d_in[2];   // (384,)
    const float* vq[3] = { (const float*)d_in[3], (const float*)d_in[4],
                           (const float*)d_in[5] };

    char* ws = (char*)d_ws;
    float*  fbuf   = (float*)(ws);                       // N*C f32   = 50331648 B
    __bf16* fbf    = (__bf16*)(ws + 50331648);           // N*C bf16  = 25165824 B
    __bf16* codebf = (__bf16*)(ws + 75497472);           // 3*K*C bf16=  4718592 B
    __bf16* wbf    = (__bf16*)(ws + 80216064);           // C*C bf16  =   294912 B
    float*  cn     = (float*)(ws + 80510976);            // 3*K f32   =    24576 B
    int*    idx    = (int*)(ws + 80535552);              // N i32     =   131072 B
    float*  loss   = (float*)(ws + 80666624);            // 3 f32

    float* out = (float*)d_out;                 // N*C quantized sum + 1 loss

    k_zero_losses<<<1, 32, 0, stream>>>(loss);
    k_conv_w<<<(CCH * CCH) / 256, 256, 0, stream>>>(wproj, wbf);
    k_prep_code<<<3 * KCB, 128, 0, stream>>>(vq[0], vq[1], vq[2], codebf, cn);

    // projection: N/128 workgroups of 8 waves
    k_proj<<<NPIX / 128, 256, 0, stream>>>(feat, bproj, wbf, fbuf, fbf);

    for (int s = 0; s < 3; ++s) {
        k_argmin<<<NPIX / 128, 256, 0, stream>>>(
            fbf, codebf + (size_t)s * KCB * CCH, cn + (size_t)s * KCB, idx);
        k_residual<<<(NPIX * (CCH / 4)) / 256, 256, 0, stream>>>(
            fbuf, fbf, vq[s], idx, out, loss + s, s == 0 ? 1 : 0);
    }

    k_final<<<1, 32, 0, stream>>>(loss, out + (size_t)NPIX * CCH);
}
